// encoder_block_11493332484792
// MI455X (gfx1250) — matrix operand order repaired and checked by reading, not compiled, not run on hardware
//
#include <hip/hip_runtime.h>
#include <hip/hip_bf16.h>

// ---------------- types & helpers ----------------
typedef __bf16 bf16;
typedef __attribute__((ext_vector_type(8)))  __bf16 v8bf;
typedef __attribute__((ext_vector_type(16))) __bf16 v16bf;
typedef __attribute__((ext_vector_type(8)))  float  v8f;
typedef int v4i_ __attribute__((vector_size(16)));   // matches builtin param type from diagnostic

#define DEVINL static __device__ __forceinline__

constexpr int CB = 4, CS = 2048, CD = 1024, CH = 16, CHD = 64, CFF = 4096;

DEVINL bf16 f2bf(float f) {
  unsigned u = __builtin_bit_cast(unsigned, f);
  unsigned r = (u + 0x7FFFu + ((u >> 16) & 1u)) >> 16;   // round-to-nearest-even
  unsigned short s = (unsigned short)r;
  return __builtin_bit_cast(bf16, s);
}

DEVINL v8bf ld8(const bf16* p) { return *reinterpret_cast<const v8bf*>(p); }
DEVINL v16bf cat16(v8bf a, v8bf b) {
  return __builtin_shufflevector(a, b, 0,1,2,3,4,5,6,7,8,9,10,11,12,13,14,15);
}
// A fragment (row-major A): lane holds row (lane&15); 16 contiguous K at hi*16
DEVINL v16bf ldA(const bf16* p) { return cat16(ld8(p), ld8(p + 8)); }
// B fragment (Bt = N x K row-major): lane holds col (lane&15); 16 contiguous K at hi*16
DEVINL v16bf ldB(const bf16* p) { return cat16(ld8(p), ld8(p + 8)); }

DEVINL v8f wmma_bf16(v16bf a, v16bf b, v8f c) {
  return __builtin_amdgcn_wmma_f32_16x16x32_bf16(false, a, false, b, (short)0, c, false, false);
}

DEVINL float redmax16(float v) {     // max across the 16-lane half-group
  v = fmaxf(v, __shfl_xor(v, 1, 32));
  v = fmaxf(v, __shfl_xor(v, 2, 32));
  v = fmaxf(v, __shfl_xor(v, 4, 32));
  v = fmaxf(v, __shfl_xor(v, 8, 32));
  return v;
}
DEVINL float redsum16(float v) {
  v += __shfl_xor(v, 1, 32);
  v += __shfl_xor(v, 2, 32);
  v += __shfl_xor(v, 4, 32);
  v += __shfl_xor(v, 8, 32);
  return v;
}

// ---------------- async global->LDS staging (CDNA5 path with fallback) ----------------
#if __has_builtin(__builtin_amdgcn_global_load_async_to_lds_b128)
#define HAVE_ASYNC_LDS 1
// copies 2 x 16B (one 32B half-row of the B panel) per thread
#define FILL_PANEL(gp, lp)                                                          \
  do {                                                                              \
    __builtin_amdgcn_global_load_async_to_lds_b128((v4i_*)(gp), (v4i_*)(lp), 0, 0); \
    __builtin_amdgcn_global_load_async_to_lds_b128((v4i_*)((gp) + 8),               \
                                                   (v4i_*)((lp) + 8), 0, 0);        \
  } while (0)
#define WAIT_ASYNC(n) asm volatile("s_wait_asynccnt %0" ::"i"(n) : "memory")
#else
#define HAVE_ASYNC_LDS 0
#define FILL_PANEL(gp, lp)                                                           \
  do {                                                                               \
    *reinterpret_cast<v8bf*>(lp) = *reinterpret_cast<const v8bf*>(gp);               \
    *reinterpret_cast<v8bf*>((lp) + 8) = *reinterpret_cast<const v8bf*>((gp) + 8);   \
  } while (0)
#define WAIT_ASYNC(n)
#endif

// ---------------- cast / repack kernels ----------------
__global__ __launch_bounds__(256) void cast_bf16_k(const float* __restrict__ s,
                                                   bf16* __restrict__ d, size_t n) {
  size_t i = (size_t)blockIdx.x * 256 + threadIdx.x;
  size_t st = (size_t)gridDim.x * 256;
  for (; i < n; i += st) d[i] = f2bf(s[i]);
}

// src: K x N (row-major fp32) -> dst: N x K (row-major bf16)  [LDS-tiled transpose]
__global__ __launch_bounds__(256) void transpose_cast_k(const float* __restrict__ src,
                                                        bf16* __restrict__ dst,
                                                        int K, int N) {
  __shared__ float tile[32][33];
  int n0 = blockIdx.x * 32, k0 = blockIdx.y * 32;
  int tx = threadIdx.x, ty = threadIdx.y;   // block (32, 8)
  #pragma unroll
  for (int i = 0; i < 4; ++i)
    tile[ty + i * 8][tx] = src[(size_t)(k0 + ty + i * 8) * N + n0 + tx];
  __syncthreads();
  #pragma unroll
  for (int i = 0; i < 4; ++i)
    dst[(size_t)(n0 + ty + i * 8) * K + k0 + tx] = f2bf(tile[tx][ty + i * 8]);
}

// vt[((b*H+h)*HD+d)*S + s] = vb[(b*S+s)*D + h*HD + d]
__global__ __launch_bounds__(256) void repack_v_k(const bf16* __restrict__ vb,
                                                  bf16* __restrict__ vt) {
  size_t i = (size_t)blockIdx.x * 256 + threadIdx.x;
  size_t st = (size_t)gridDim.x * 256;
  size_t n = (size_t)CB * CH * CHD * CS;
  for (; i < n; i += st) {
    int s = (int)(i % CS);
    size_t rem = i / CS;
    int d = (int)(rem % CHD); rem /= CHD;
    int h = (int)(rem % CH);
    int b = (int)(rem / CH);
    vt[i] = vb[(size_t)(b * CS + s) * CD + h * CHD + d];
  }
}

// ---------------- WMMA GEMM: C[M,N] = A[M,K](bf16) * Bt[N,K](bf16) ----------------
// Block tile 128(M) x 128(N); 8 waves as 4x2; wave tile 32(M) x 64(N).
// B panel (128 x 32) double-buffered through LDS (async->LDS when available).
// EPI: 0 = store bf16 ; 1 = store fp32 ; 2 = +bias, ReLU, store bf16 ; 3 = +bias, store fp32
template <int EPI>
__global__ __launch_bounds__(256) void gemm_bf16_k(const bf16* __restrict__ A,
                                                   const bf16* __restrict__ Bt,
                                                   float* __restrict__ Cf,
                                                   bf16* __restrict__ Cb,
                                                   const float* __restrict__ bias,
                                                   int M, int N, int K) {
  constexpr int BST = 40;                       // LDS row stride (bf16): 80B, conflict-free, 16B-aligned
  __shared__ bf16 Bs[2][128 * BST];
  int tid = threadIdx.x, lane = tid & 31, w = tid >> 5;
  int wm = w >> 1, wn = w & 1;                  // 4 x 2 waves
  int m0 = blockIdx.y * 128 + wm * 32;
  int nblk = blockIdx.x * 128;
  int nloc = wn * 64;
  int ml = lane & 15, hi = lane >> 4;
  (void)M;

  // B-panel fill indexing: 256 threads x 32B = 128 rows x 64B
  int frow = tid >> 1;
  int fcol = (tid & 1) * 16;                    // element offset of this thread's 32B half-row
  const bf16* gfill = Bt + (size_t)(nblk + frow) * K + fcol;
  bf16* lfill0 = &Bs[0][frow * BST + fcol];
  bf16* lfill1 = &Bs[1][frow * BST + fcol];

  v8f acc[2][4];
  #pragma unroll
  for (int i = 0; i < 2; ++i)
    #pragma unroll
    for (int j = 0; j < 4; ++j) { v8f z = {0,0,0,0,0,0,0,0}; acc[i][j] = z; }

  const bf16* pa0 = A + (size_t)(m0 + ml) * K;
  const bf16* pa1 = A + (size_t)(m0 + 16 + ml) * K;

  const int nk = K >> 5;
  FILL_PANEL(gfill, lfill0);                    // prefill buffer 0 (k-step 0)

  for (int ks = 0; ks < nk; ++ks) {
    int cur = ks & 1;
    if (ks + 1 < nk) {                          // uniform branch: EXEC stays all-ones
      FILL_PANEL(gfill + (size_t)(ks + 1) * 32, (cur == 0) ? lfill1 : lfill0);
      WAIT_ASYNC(2);                            // oldest 2 (current panel) complete
    } else {
      WAIT_ASYNC(0);
    }
    __syncthreads();

    int k0 = ks * 32;
    __builtin_prefetch(pa0 + k0 + 128, 0, 1);   // global_prefetch_b8 on the A stream
    v16bf a0 = ldA(pa0 + k0 + hi * 16);
    v16bf a1 = ldA(pa1 + k0 + hi * 16);
    v16bf bfr[4];
    #pragma unroll
    for (int j = 0; j < 4; ++j)
      bfr[j] = ldB(&Bs[cur][(nloc + j * 16 + ml) * BST + hi * 16]);
    #pragma unroll
    for (int j = 0; j < 4; ++j) {
      acc[0][j] = wmma_bf16(a0, bfr[j], acc[0][j]);
      acc[1][j] = wmma_bf16(a1, bfr[j], acc[1][j]);
    }
    __syncthreads();
  }

  #pragma unroll
  for (int i = 0; i < 2; ++i)
    #pragma unroll
    for (int j = 0; j < 4; ++j)
      #pragma unroll
      for (int r = 0; r < 8; ++r) {
        size_t row = (size_t)(m0 + i * 16 + hi * 8 + r);
        size_t col = (size_t)(nblk + nloc + j * 16 + ml);
        float v = acc[i][j][r];
        if (EPI == 0) {
          Cb[row * N + col] = f2bf(v);
        } else if (EPI == 1) {
          Cf[row * N + col] = v;
        } else if (EPI == 2) {
          v += bias[col];
          Cb[row * N + col] = f2bf(fmaxf(v, 0.0f));
        } else {
          v += bias[col];
          Cf[row * N + col] = v;
        }
      }
}

// ---------------- flash attention: one wave = 16 queries x all valid keys ----------------
__global__ __launch_bounds__(256) void attn_k(const bf16* __restrict__ qb,
                                              const bf16* __restrict__ kb,
                                              const bf16* __restrict__ vt,
                                              const int* __restrict__ valid_lens,
                                              bf16* __restrict__ attn_out) {
  __shared__ bf16 psh[8][16 * 40];                   // stride 40 bf16: conflict-free, 16B-aligned
  int tid = threadIdx.x, lane = tid & 31, w = tid >> 5;
  int gw = blockIdx.x * 8 + w;
  int qt = gw & (CS / 16 - 1);                       // 128 query tiles per (b,h)
  int bh = gw >> 7;
  int h = bh & (CH - 1);
  int b = bh >> 4;
  int ml = lane & 15, hi = lane >> 4;
  int q0 = qt * 16;
  int vl = valid_lens[b];
  int nkt = (vl + 31) >> 5;                          // tiles fully past valid_len contribute exactly 0

  const bf16* qrow = qb + (size_t)(b * CS + q0 + ml) * CD + h * CHD;
  v16bf qf0 = cat16(ld8(qrow + hi * 16),      ld8(qrow + 8 + hi * 16));
  v16bf qf1 = cat16(ld8(qrow + 32 + hi * 16), ld8(qrow + 40 + hi * 16));

  float mrow[8], lrow[8];
  v8f o[4];
  #pragma unroll
  for (int r = 0; r < 8; ++r) { mrow[r] = -1e30f; lrow[r] = 0.0f; }
  #pragma unroll
  for (int j = 0; j < 4; ++j) { v8f z = {0,0,0,0,0,0,0,0}; o[j] = z; }

  for (int kt = 0; kt < nkt; ++kt) {
    int kbase = kt * 32;
    v8f s0 = {0,0,0,0,0,0,0,0}, s1 = s0;
    {
      const bf16* kr0 = kb + (size_t)(b * CS + kbase + ml) * CD + h * CHD + hi * 16;
      const bf16* kr1 = kb + (size_t)(b * CS + kbase + 16 + ml) * CD + h * CHD + hi * 16;
      s0 = wmma_bf16(qf0, ldB(kr0), s0);
      s0 = wmma_bf16(qf1, ldB(kr0 + 32), s0);
      s1 = wmma_bf16(qf0, ldB(kr1), s1);
      s1 = wmma_bf16(qf1, ldB(kr1 + 32), s1);
    }
    int key0 = kbase + ml, key1 = kbase + 16 + ml;
    bool ok0 = key0 < vl, ok1 = key1 < vl;
    float alpha[8];
    #pragma unroll
    for (int r = 0; r < 8; ++r) {
      float v0 = s0[r] * 0.125f;                     // 1/sqrt(HD)
      float v1 = s1[r] * 0.125f;
      float tm = fmaxf(ok0 ? v0 : -1e30f, ok1 ? v1 : -1e30f);
      tm = redmax16(tm);
      float mn = fmaxf(mrow[r], tm);
      float a = __expf(mrow[r] - mn);
      float p0 = ok0 ? __expf(v0 - mn) : 0.0f;
      float p1 = ok1 ? __expf(v1 - mn) : 0.0f;
      lrow[r] = lrow[r] * a + redsum16(p0 + p1);
      mrow[r] = mn;
      alpha[r] = a;
      psh[w][(hi * 8 + r) * 40 + ml]      = f2bf(p0);   // C-layout -> LDS
      psh[w][(hi * 8 + r) * 40 + 16 + ml] = f2bf(p1);
    }
    #pragma unroll
    for (int j = 0; j < 4; ++j)
      #pragma unroll
      for (int r = 0; r < 8; ++r) o[j][r] *= alpha[r];

    // P (16x32) back as A fragment (same-wave LDS RAW, DScnt-ordered)
    const bf16* pr = &psh[w][ml * 40 + hi * 16];
    v16bf pfrag = cat16(ld8(pr), ld8(pr + 8));
    #pragma unroll
    for (int j = 0; j < 4; ++j) {
      const bf16* vr = vt + (size_t)((b * CH + h) * CHD + j * 16 + ml) * CS + kbase + hi * 16;
      o[j] = wmma_bf16(pfrag, ldB(vr), o[j]);
    }
  }

  float inv[8];
  #pragma unroll
  for (int r = 0; r < 8; ++r) inv[r] = 1.0f / lrow[r];
  #pragma unroll
  for (int j = 0; j < 4; ++j)
    #pragma unroll
    for (int r = 0; r < 8; ++r) {
      size_t row = (size_t)(b * CS + q0 + hi * 8 + r);
      attn_out[row * CD + h * CHD + j * 16 + ml] = f2bf(o[j][r] * inv[r]);
    }
}

// ---------------- LayerNorm(a + res), one block per row of D=1024 ----------------
__global__ __launch_bounds__(256) void ln_k(const float* __restrict__ a,
                                            const float* __restrict__ res,
                                            const float* __restrict__ g,
                                            const float* __restrict__ be,
                                            float* __restrict__ outf,
                                            bf16* __restrict__ outb) {
  int row = blockIdx.x, tid = threadIdx.x;
  __shared__ float s1[256], s2[256];
  const float* pa = a + (size_t)row * CD;
  const float* pr = res + (size_t)row * CD;
  float v[4], ls = 0.f, lq = 0.f;
  #pragma unroll
  for (int i = 0; i < 4; ++i) {
    int c = tid + i * 256;
    float x = pa[c] + pr[c];
    v[i] = x; ls += x; lq += x * x;
  }
  s1[tid] = ls; s2[tid] = lq;
  __syncthreads();
  for (int st = 128; st > 0; st >>= 1) {
    if (tid < st) { s1[tid] += s1[tid + st]; s2[tid] += s2[tid + st]; }
    __syncthreads();
  }
  float mu = s1[0] * (1.0f / CD);
  float var = s2[0] * (1.0f / CD) - mu * mu;
  float rs = rsqrtf(var + 1e-5f);
  #pragma unroll
  for (int i = 0; i < 4; ++i) {
    int c = tid + i * 256;
    float y = (v[i] - mu) * rs * g[c] + be[c];
    if (outf) outf[(size_t)row * CD + c] = y;
    if (outb) outb[(size_t)row * CD + c] = f2bf(y);
  }
}

// ---------------- launcher ----------------
extern "C" void kernel_launch(void* const* d_in, const int* in_sizes, int n_in,
                              void* d_out, int out_size, void* d_ws, size_t ws_size,
                              hipStream_t stream) {
  (void)in_sizes; (void)n_in; (void)out_size; (void)ws_size;
  const float* x   = (const float*)d_in[0];
  const int*   vl  = (const int*)d_in[1];
  const float* Wq  = (const float*)d_in[2];
  const float* Wk  = (const float*)d_in[3];
  const float* Wv  = (const float*)d_in[4];
  const float* Wo  = (const float*)d_in[5];
  const float* W1  = (const float*)d_in[6];
  const float* b1  = (const float*)d_in[7];
  const float* W2  = (const float*)d_in[8];
  const float* b2  = (const float*)d_in[9];
  const float* g1  = (const float*)d_in[10];
  const float* be1 = (const float*)d_in[11];
  const float* g2  = (const float*)d_in[12];
  const float* be2 = (const float*)d_in[13];
  float* out = (float*)d_out;

  char* ws = (char*)d_ws;
  const size_t MB = 1ull << 20;
  bf16*  xb    = (bf16*)(ws + 0 * MB);     // 16 MB
  bf16*  wqt   = (bf16*)(ws + 16 * MB);    // 2 MB each
  bf16*  wkt   = (bf16*)(ws + 18 * MB);
  bf16*  wvt   = (bf16*)(ws + 20 * MB);
  bf16*  wot   = (bf16*)(ws + 22 * MB);
  bf16*  w1t   = (bf16*)(ws + 24 * MB);    // 8 MB
  bf16*  w2t   = (bf16*)(ws + 32 * MB);    // 8 MB
  bf16*  qb    = (bf16*)(ws + 40 * MB);    // 16 MB
  bf16*  kb    = (bf16*)(ws + 56 * MB);    // 16 MB (reused as res1 bf16 after attention)
  bf16*  vb    = (bf16*)(ws + 72 * MB);    // 16 MB (reused as attn bf16 after vt repack)
  bf16*  vt    = (bf16*)(ws + 88 * MB);    // 16 MB
  float* gout  = (float*)(ws + 104 * MB);  // 32 MB fp32 (attn_out, then ff_out)
  float* res1f = (float*)(ws + 136 * MB);  // 32 MB fp32
  bf16*  hbf   = (bf16*)(ws + 168 * MB);   // 64 MB  (total 232 MB)
  bf16*  attn_bf = vb;
  bf16*  r1b     = kb;

  dim3 blk(256);
  dim3 tblk(32, 8);
  const int Mrows = CB * CS;               // 8192

  cast_bf16_k<<<4096, blk, 0, stream>>>(x, xb, (size_t)CB * CS * CD);
  transpose_cast_k<<<dim3(CD / 32, CD / 32), tblk, 0, stream>>>(Wq, wqt, CD, CD);
  transpose_cast_k<<<dim3(CD / 32, CD / 32), tblk, 0, stream>>>(Wk, wkt, CD, CD);
  transpose_cast_k<<<dim3(CD / 32, CD / 32), tblk, 0, stream>>>(Wv, wvt, CD, CD);
  transpose_cast_k<<<dim3(CD / 32, CD / 32), tblk, 0, stream>>>(Wo, wot, CD, CD);
  transpose_cast_k<<<dim3(CFF / 32, CD / 32), tblk, 0, stream>>>(W1, w1t, CD, CFF);
  transpose_cast_k<<<dim3(CD / 32, CFF / 32), tblk, 0, stream>>>(W2, w2t, CFF, CD);

  gemm_bf16_k<0><<<dim3(CD / 128, Mrows / 128), blk, 0, stream>>>(xb, wqt, nullptr, qb, nullptr, Mrows, CD, CD);
  gemm_bf16_k<0><<<dim3(CD / 128, Mrows / 128), blk, 0, stream>>>(xb, wkt, nullptr, kb, nullptr, Mrows, CD, CD);
  gemm_bf16_k<0><<<dim3(CD / 128, Mrows / 128), blk, 0, stream>>>(xb, wvt, nullptr, vb, nullptr, Mrows, CD, CD);

  repack_v_k<<<4096, blk, 0, stream>>>(vb, vt);

  attn_k<<<dim3(CB * CH * (CS / 16) / 8), blk, 0, stream>>>(qb, kb, vt, vl, attn_bf);

  gemm_bf16_k<1><<<dim3(CD / 128, Mrows / 128), blk, 0, stream>>>(attn_bf, wot, gout, nullptr, nullptr, Mrows, CD, CD);
  ln_k<<<Mrows, blk, 0, stream>>>(gout, x, g1, be1, res1f, r1b);

  gemm_bf16_k<2><<<dim3(CFF / 128, Mrows / 128), blk, 0, stream>>>(r1b, w1t, nullptr, hbf, b1, Mrows, CFF, CD);
  gemm_bf16_k<3><<<dim3(CD / 128, Mrows / 128), blk, 0, stream>>>(hbf, w2t, gout, nullptr, b2, Mrows, CD, CFF);
  ln_k<<<Mrows, blk, 0, stream>>>(gout, res1f, g2, be2, out, nullptr);
}
